// RecursionLatticeLayer_16071767622086
// MI455X (gfx1250) — compile-verified, matrix operand order checked by reading, not run on hardware
//
#include <hip/hip_runtime.h>

// ---------------------------------------------------------------------------
// MI455X (gfx1250) fused transformer layer, bf16 WMMA pipeline.
// All GEMMs: A(bf16) x B(fp32 weights or bf16 activations) -> fp32 acc via
// v_wmma_f32_16x16x32_bf16, LDS-staged 128x128x32 tiles, 8 waves/block.
// Workspace use ~320 MB (scores H*T*T bf16 dominate).
// ---------------------------------------------------------------------------

typedef unsigned int uint;
typedef unsigned short ushort;
typedef __attribute__((ext_vector_type(16))) __bf16 v16bf;
typedef __attribute__((ext_vector_type(8)))  float  v8f;

#define DEVI __device__ __forceinline__

DEVI ushort f2bf(float f) {
  uint u = __builtin_bit_cast(uint, f);
  u += 0x7fffu + ((u >> 16) & 1u);           // round-to-nearest-even
  return (ushort)(u >> 16);
}
DEVI float bf2f(ushort h) {
  uint u = ((uint)h) << 16;
  return __builtin_bit_cast(float, u);
}
// pack two fp32 -> packed bf16 dword (v_cvt_pk_bf16_f32 when available)
#if __has_builtin(__builtin_amdgcn_cvt_pk_bf16_f32)
DEVI uint pack2bf(float lo, float hi) {
  auto r = __builtin_amdgcn_cvt_pk_bf16_f32(lo, hi);
  return __builtin_bit_cast(uint, r);
}
#else
DEVI uint pack2bf(float lo, float hi) {
  return (uint)f2bf(lo) | ((uint)f2bf(hi) << 16);
}
#endif

union Frag16 { v16bf v; uint4 q[2]; };

// ---------------------------------------------------------------------------
// WMMA GEMM:  C[M,N] = epilogue( scale * A[M,K] @ B[K,N] )
//   BT:  B stored [N,K] row-major (i.e. we compute A @ B_storedᵀ)  (for Q@Kᵀ)
//   BB16: B elements are bf16 (activations) instead of fp32 (weights)
// LDS holds A tile [128][32] and B tile transposed [n][k] so both fragment
// loads are 32 contiguous bytes per lane (k = 16*(lane/16) + s, same order for A and B).
// ---------------------------------------------------------------------------
template <bool BT, bool BB16>
__global__ __launch_bounds__(256) void gemm_wmma(
    const ushort* __restrict__ A, int lda, long long aZ,
    const void*  __restrict__ Bp, int ldb, long long bZ,
    float*  __restrict__ outF, int ldcF, long long cZF,
    ushort* __restrict__ outH, int ldcH, long long cZH,
    const float* __restrict__ bias,
    const float* __restrict__ res,    // fp32 residual added pre-store (ldcF)
    float scale, int act,             // act: 0=none, 1=exact GELU
    int M, int N, int K)
{
  __shared__ ushort ldsA[128 * 40];   // 40-elem row stride (pad vs conflicts)
  __shared__ ushort ldsB[128 * 40];   // stored [n][k]

  const int z = blockIdx.z;
  A += (size_t)z * (size_t)aZ;
  const float*  Bf = (const float*)Bp  + (BB16 ? 0 : (size_t)z * (size_t)bZ);
  const ushort* Bh = (const ushort*)Bp + (BB16 ? (size_t)z * (size_t)bZ : 0);

  const int bn   = blockIdx.x * 128;
  const int bm   = blockIdx.y * 128;
  const int tid  = threadIdx.x;
  const int lane = tid & 31;
  const int wave = tid >> 5;          // wave32: 8 waves / 256 threads
  const int wm   = (wave >> 1) * 32;  // 4 waves along M
  const int wn   = (wave & 1) * 64;   // 2 waves along N

  const bool fullN = (bn + 128 <= N); // block-uniform

  v8f acc[2][4] = {};

  for (int kt = 0; kt < K; kt += 32) {
    // ---- stage A tile [128 rows][32 k] (bf16, contiguous) ----
    {
      const int r  = tid >> 1;
      const int c0 = (tid & 1) * 16;
      const ushort* src = A + (size_t)(bm + r) * lda + kt + c0;
      uint4 q0 = *(const uint4*)src;
      uint4 q1 = *(const uint4*)(src + 8);
      *(uint4*)&ldsA[r * 40 + c0]     = q0;
      *(uint4*)&ldsA[r * 40 + c0 + 8] = q1;
      if (kt + 32 < K) __builtin_prefetch(src + 32, 0, 1);  // global_prefetch
    }
    // ---- stage B tile as [n][k] ----
    if constexpr (BT) {
      // global B is [N,K] row-major (bf16): straight copy, k contiguous
      const int n  = tid >> 1;
      const int k0 = (tid & 1) * 16;
      const int gn = bn + n;
      uint4 q0 = {0, 0, 0, 0}, q1 = {0, 0, 0, 0};
      if (gn < N) {
        const ushort* src = Bh + (size_t)gn * ldb + kt + k0;
        q0 = *(const uint4*)src;
        q1 = *(const uint4*)(src + 8);
      }
      *(uint4*)&ldsB[n * 40 + k0]     = q0;
      *(uint4*)&ldsB[n * 40 + k0 + 8] = q1;
    } else if (fullN) {
      // fast path: thread covers K-pair x 8 N, vector loads, packed b32 LDS
      const int k0 = (tid & 15) * 2;        // 0,2,..,30
      const int n0 = (tid >> 4) * 8;        // 0,8,..,120
      const int gk = kt + k0;
      if constexpr (BB16) {
        uint4 r0 = *(const uint4*)(Bh + (size_t)gk * ldb + bn + n0);
        uint4 r1 = *(const uint4*)(Bh + (size_t)(gk + 1) * ldb + bn + n0);
        const ushort* h0 = (const ushort*)&r0;
        const ushort* h1 = (const ushort*)&r1;
#pragma unroll
        for (int j = 0; j < 8; ++j) {
          const uint packed = (uint)h0[j] | ((uint)h1[j] << 16);
          *(uint*)&ldsB[(n0 + j) * 40 + k0] = packed;
        }
      } else {
        const float4* p0 = (const float4*)(Bf + (size_t)gk * ldb + bn + n0);
        const float4* p1 = (const float4*)(Bf + (size_t)(gk + 1) * ldb + bn + n0);
        const float4 a0 = p0[0], a1 = p0[1];
        const float4 b0 = p1[0], b1 = p1[1];
        const float r0[8] = {a0.x, a0.y, a0.z, a0.w, a1.x, a1.y, a1.z, a1.w};
        const float r1[8] = {b0.x, b0.y, b0.z, b0.w, b1.x, b1.y, b1.z, b1.w};
#pragma unroll
        for (int j = 0; j < 8; ++j)
          *(uint*)&ldsB[(n0 + j) * 40 + k0] = pack2bf(r0[j], r1[j]);
      }
    } else {
      // guarded scalar fallback (only the rank-32 LoRA GEMM hits this)
      const int kk = tid >> 3;             // 0..31
      const int n0 = (tid & 7) * 16;       // 0..112
      const int gk = kt + kk;
#pragma unroll
      for (int i = 0; i < 16; ++i) {
        const int gn = bn + n0 + i;
        float v = 0.f;
        if (gn < N) {
          if constexpr (BB16) v = bf2f(Bh[(size_t)gk * ldb + gn]);
          else                v = Bf[(size_t)gk * ldb + gn];
        }
        ldsB[(n0 + i) * 40 + kk] = f2bf(v);
      }
    }
    __syncthreads();

    // ---- fragments: lane L -> row/col (L&15), K-half (L>>4)*16 ----
    Frag16 fa[2], fb[4];
    const int rc = lane & 15;
    const int ko = (lane >> 4) * 16;
#pragma unroll
    for (int mi = 0; mi < 2; ++mi) {
      const ushort* p = &ldsA[(wm + mi * 16 + rc) * 40 + ko];
      fa[mi].q[0] = *(const uint4*)p;
      fa[mi].q[1] = *(const uint4*)(p + 8);
    }
#pragma unroll
    for (int ni = 0; ni < 4; ++ni) {
      const ushort* p = &ldsB[(wn + ni * 16 + rc) * 40 + ko];
      fb[ni].q[0] = *(const uint4*)p;
      fb[ni].q[1] = *(const uint4*)(p + 8);
    }
#pragma unroll
    for (int mi = 0; mi < 2; ++mi)
#pragma unroll
      for (int ni = 0; ni < 4; ++ni)
        acc[mi][ni] = __builtin_amdgcn_wmma_f32_16x16x32_bf16(
            false, fa[mi].v, false, fb[ni].v, (short)0, acc[mi][ni],
            false, false);
    __syncthreads();
  }

  // ---- epilogue; C layout: VGPR g holds row g + 8*(lane>=16), col lane&15 --
  const int rowHalf = (lane >> 4) << 3;
#pragma unroll
  for (int mi = 0; mi < 2; ++mi) {
#pragma unroll
    for (int ni = 0; ni < 4; ++ni) {
      const int col = bn + wn + ni * 16 + (lane & 15);
      if (col >= N) continue;
#pragma unroll
      for (int g = 0; g < 8; ++g) {
        const int row = bm + wm + mi * 16 + rowHalf + g;
        float v = acc[mi][ni][g] * scale;
        if (bias) v += bias[col];
        if (act == 1) v = 0.5f * v * (1.0f + erff(v * 0.70710678118654752f));
        if (res)  v += res[(size_t)row * ldcF + col];
        if (outF) outF[(size_t)z * cZF + (size_t)row * ldcF + col] = v;
        if (outH) outH[(size_t)z * cZH + (size_t)row * ldcH + col] = f2bf(v);
      }
    }
  }
}

// ---------------------------------------------------------------------------
// LayerNorm over D=2048, one block (8 waves) per row.
// ---------------------------------------------------------------------------
__global__ __launch_bounds__(256) void layernorm_k(
    const float* __restrict__ x, const float* __restrict__ g,
    const float* __restrict__ b, float* __restrict__ outF,
    ushort* __restrict__ outH, int outLd, int D)
{
  __shared__ float red[256];
  const int tid = threadIdx.x;
  const float* xr = x + (size_t)blockIdx.x * D;
  float v[8];
  float s = 0.f;
#pragma unroll
  for (int i = 0; i < 8; ++i) { v[i] = xr[tid + i * 256]; s += v[i]; }
  red[tid] = s;
  __syncthreads();
  for (int o = 128; o > 0; o >>= 1) {
    if (tid < o) red[tid] += red[tid + o];
    __syncthreads();
  }
  const float mean = red[0] / (float)D;
  __syncthreads();
  s = 0.f;
#pragma unroll
  for (int i = 0; i < 8; ++i) { float d = v[i] - mean; s += d * d; }
  red[tid] = s;
  __syncthreads();
  for (int o = 128; o > 0; o >>= 1) {
    if (tid < o) red[tid] += red[tid + o];
    __syncthreads();
  }
  const float rstd = rsqrtf(red[0] / (float)D + 1e-5f);
#pragma unroll
  for (int i = 0; i < 8; ++i) {
    const int c = tid + i * 256;
    const float o = (v[i] - mean) * rstd * g[c] + b[c];
    if (outF) outF[(size_t)blockIdx.x * outLd + c] = o;
    if (outH) outH[(size_t)blockIdx.x * outLd + c] = f2bf(o);
  }
}

// ---------------------------------------------------------------------------
// In-place masked row softmax on bf16 scores; one block per row (len T=2048).
// ---------------------------------------------------------------------------
__global__ __launch_bounds__(256) void softmax_k(
    ushort* __restrict__ S, const int* __restrict__ mask, int T)
{
  __shared__ float red[256];
  const int tid = threadIdx.x;
  ushort* row = S + (size_t)blockIdx.x * T;
  float v[8];
  float m = -3.0e38f;
#pragma unroll
  for (int i = 0; i < 8; ++i) {
    const int c = tid + i * 256;
    float x = bf2f(row[c]);
    if (mask[c] == 0) x = -3.0e38f;
    v[i] = x;
    m = fmaxf(m, x);
  }
  red[tid] = m;
  __syncthreads();
  for (int o = 128; o > 0; o >>= 1) {
    if (tid < o) red[tid] = fmaxf(red[tid], red[tid + o]);
    __syncthreads();
  }
  m = red[0];
  __syncthreads();
  float s = 0.f;
#pragma unroll
  for (int i = 0; i < 8; ++i) { v[i] = __expf(v[i] - m); s += v[i]; }
  red[tid] = s;
  __syncthreads();
  for (int o = 128; o > 0; o >>= 1) {
    if (tid < o) red[tid] += red[tid + o];
    __syncthreads();
  }
  const float inv = 1.0f / red[0];
#pragma unroll
  for (int i = 0; i < 8; ++i) row[tid + i * 256] = f2bf(v[i] * inv);
}

// fp32 -> bf16 strided copy (for building concat inputs)
__global__ void cvt_k(const float* __restrict__ src, int sld,
                      ushort* __restrict__ dst, int dld, int cols)
{
  const float* s = src + (size_t)blockIdx.x * sld;
  ushort* d = dst + (size_t)blockIdx.x * dld;
  for (int c = threadIdx.x; c < cols; c += blockDim.x) d[c] = f2bf(s[c]);
}

// residual2 = hs + attn_out + loop_scale * sigmoid(gpre) * lora
__global__ void combine_k(const float* __restrict__ hs,
                          const float* __restrict__ ao,
                          const float* __restrict__ gpre,
                          const float* __restrict__ lora,
                          const float* __restrict__ ls,
                          float* __restrict__ res2, int n)
{
  const float scale = *ls;
  for (int i = blockIdx.x * blockDim.x + threadIdx.x; i < n;
       i += gridDim.x * blockDim.x) {
    const float gate = 1.0f / (1.0f + __expf(-gpre[i]));
    res2[i] = hs[i] + ao[i] + scale * gate * lora[i];
  }
}

// ---------------------------------------------------------------------------
extern "C" void kernel_launch(void* const* d_in, const int* in_sizes, int n_in,
                              void* d_out, int out_size, void* d_ws,
                              size_t ws_size, hipStream_t stream)
{
  (void)in_sizes; (void)n_in; (void)out_size; (void)ws_size;
  constexpr int T = 2048, D = 2048, H = 16, HD = 128, FFN = 8192, RANK = 32;
  constexpr long long TD = (long long)T * D;

  const float* hs    = (const float*)d_in[0];
  const float* past1 = (const float*)d_in[1];
  const float* past2 = (const float*)d_in[2];
  const int*   maskp = (const int*)d_in[3];
  const float* Wq    = (const float*)d_in[4];
  const float* Wk    = (const float*)d_in[5];
  const float* Wv    = (const float*)d_in[6];
  const float* Wo    = (const float*)d_in[7];
  const float* Wgate = (const float*)d_in[8];
  const float* Wld   = (const float*)d_in[9];
  const float* Wlu   = (const float*)d_in[10];
  const float* lsp   = (const float*)d_in[11];
  const float* Wf1   = (const float*)d_in[12];
  const float* bf1   = (const float*)d_in[13];
  const float* Wf2   = (const float*)d_in[14];
  const float* bf2   = (const float*)d_in[15];
  const float* g1 = (const float*)d_in[16]; const float* b1 = (const float*)d_in[17];
  const float* g2 = (const float*)d_in[18]; const float* b2 = (const float*)d_in[19];
  const float* g3 = (const float*)d_in[20]; const float* b3 = (const float*)d_in[21];

  float* out = (float*)d_out;            // [0:TD]=output, [TD:2TD]=residual2

  // ---- workspace bump allocator (256B aligned) ----
  char* wsp = (char*)d_ws;
  size_t off = 0;
  auto alloc = [&](size_t bytes) -> void* {
    void* p = wsp + off;
    off += (bytes + 255) & ~(size_t)255;
    return p;
  };
  ushort* qin  = (ushort*)alloc((size_t)T * 3 * D * 2);   // [normed|past1|past2]
  ushort* gin  = (ushort*)alloc((size_t)T * 2 * D * 2);   // [attn_out|past2]
  ushort* Qb   = (ushort*)alloc((size_t)TD * 2);
  ushort* Kb   = (ushort*)alloc((size_t)TD * 2);
  ushort* Vb   = (ushort*)alloc((size_t)TD * 2);
  ushort* Sp   = (ushort*)alloc((size_t)H * T * T * 2);   // scores/probs, 134MB
  ushort* ctx  = (ushort*)alloc((size_t)TD * 2);
  float*  aoF  = (float*)alloc((size_t)TD * 4);
  float*  gpre = (float*)alloc((size_t)TD * 4);
  ushort* mid  = (ushort*)alloc((size_t)T * RANK * 2);
  float*  loraF= (float*)alloc((size_t)TD * 4);
  ushort* h2   = (ushort*)alloc((size_t)TD * 2);
  ushort* f1   = (ushort*)alloc((size_t)T * FFN * 2);
  float*  hid2 = (float*)alloc((size_t)TD * 4);

  const dim3 blk(256);
  const dim3 gDD(D / 128, T / 128, 1);        // [T x D] outputs
  const float qscale = 0.08838834764831845f;  // HD^-0.5

  // 1) LN1 -> bf16 into qin[:,0:D]
  layernorm_k<<<T, blk, 0, stream>>>(hs, g1, b1, nullptr, qin, 3 * D, D);
  // 2) fill concat buffers
  cvt_k<<<T, blk, 0, stream>>>(past1, D, qin + D, 3 * D, D);
  cvt_k<<<T, blk, 0, stream>>>(past2, D, qin + 2 * D, 3 * D, D);
  cvt_k<<<T, blk, 0, stream>>>(past2, D, gin + D, 2 * D, D);

  // 3) Q = (qin @ Wq) * scale      [K=3D]
  gemm_wmma<false, false><<<gDD, blk, 0, stream>>>(
      qin, 3 * D, 0, Wq, D, 0, nullptr, 0, 0, Qb, D, 0,
      nullptr, nullptr, qscale, 0, T, D, 3 * D);
  // 4) K = normed @ Wk ; V = normed @ Wv   (normed = qin[:,0:D], lda=3D)
  gemm_wmma<false, false><<<gDD, blk, 0, stream>>>(
      qin, 3 * D, 0, Wk, D, 0, nullptr, 0, 0, Kb, D, 0,
      nullptr, nullptr, 1.f, 0, T, D, D);
  gemm_wmma<false, false><<<gDD, blk, 0, stream>>>(
      qin, 3 * D, 0, Wv, D, 0, nullptr, 0, 0, Vb, D, 0,
      nullptr, nullptr, 1.f, 0, T, D, D);

  // 5) per-head scores S_h = Q_h @ K_hᵀ   (B transposed, bf16)
  {
    dim3 g(T / 128, T / 128, H);
    gemm_wmma<true, true><<<g, blk, 0, stream>>>(
        Qb, D, HD, Kb, D, HD, nullptr, 0, 0, Sp, T, (long long)T * T,
        nullptr, nullptr, 1.f, 0, T, T, HD);
  }
  // 6) masked softmax rows (H*T rows of length T), in place
  softmax_k<<<H * T, blk, 0, stream>>>(Sp, maskp, T);
  // 7) ctx_h = P_h @ V_h
  {
    dim3 g(HD / 128, T / 128, H);
    gemm_wmma<false, true><<<g, blk, 0, stream>>>(
        Sp, T, (long long)T * T, Vb, D, HD, nullptr, 0, 0, ctx, D, HD,
        nullptr, nullptr, 1.f, 0, T, HD, T);
  }
  // 8) attn_out = ctx @ Wo  (fp32 + bf16 copy into gin[:,0:D])
  gemm_wmma<false, false><<<gDD, blk, 0, stream>>>(
      ctx, D, 0, Wo, D, 0, aoF, D, 0, gin, 2 * D, 0,
      nullptr, nullptr, 1.f, 0, T, D, D);
  // 9) gate pre-activation = [attn_out|past2] @ Wgate   [K=2D]
  gemm_wmma<false, false><<<gDD, blk, 0, stream>>>(
      gin, 2 * D, 0, Wgate, D, 0, gpre, D, 0, nullptr, 0, 0,
      nullptr, nullptr, 1.f, 0, T, D, 2 * D);
  // 10) lora: mid = past2 @ Wld [N=32], lora = mid @ Wlu [K=32]
  {
    dim3 g(1, T / 128, 1);
    gemm_wmma<false, false><<<g, blk, 0, stream>>>(
        gin + D, 2 * D, 0, Wld, RANK, 0, nullptr, 0, 0, mid, RANK, 0,
        nullptr, nullptr, 1.f, 0, T, RANK, D);
  }
  gemm_wmma<false, false><<<gDD, blk, 0, stream>>>(
      mid, RANK, 0, Wlu, D, 0, loraF, D, 0, nullptr, 0, 0,
      nullptr, nullptr, 1.f, 0, T, D, RANK);

  // 11) residual2 = hs + attn_out + ls*sigmoid(gpre)*lora  -> d_out[TD:2TD]
  combine_k<<<512, blk, 0, stream>>>(hs, aoF, gpre, loraF, lsp,
                                     out + TD, (int)TD);
  // 12) FFN: h2 = LN2(residual2); f1 = gelu(h2@Wf1+bf1); hid2 = res2 + f1@Wf2+bf2
  layernorm_k<<<T, blk, 0, stream>>>(out + TD, g2, b2, nullptr, h2, D, D);
  {
    dim3 g(FFN / 128, T / 128, 1);
    gemm_wmma<false, false><<<g, blk, 0, stream>>>(
        h2, D, 0, Wf1, FFN, 0, nullptr, 0, 0, f1, FFN, 0,
        bf1, nullptr, 1.f, 1, T, FFN, D);
  }
  gemm_wmma<false, false><<<gDD, blk, 0, stream>>>(
      f1, FFN, 0, Wf2, D, 0, hid2, D, 0, nullptr, 0, 0,
      bf2, out + TD, 1.f, 0, T, D, FFN);
  // 13) output = LN3(hid2) -> d_out[0:TD]
  layernorm_k<<<T, blk, 0, stream>>>(hid2, g3, b3, out, nullptr, D, D);
}
